// GistBuffer_7834020348584
// MI455X (gfx1250) — matrix operand order repaired and checked by reading, not compiled, not run on hardware
//
#include <hip/hip_runtime.h>
#include <hip/hip_bf16.h>

typedef __attribute__((ext_vector_type(16))) _Float16 v16h;
typedef __attribute__((ext_vector_type(8)))  _Float16 v8h;
typedef __attribute__((ext_vector_type(8)))  float    v8f;

#define NEG_INF (-__builtin_inff())

// ---------------------------------------------------------------------------
// Kernel 1: Eq = (f16) exp(query)   [B*d elements]
// ---------------------------------------------------------------------------
__global__ void expq_kernel(const float* __restrict__ q,
                            _Float16* __restrict__ eq, int total) {
    int i = blockIdx.x * blockDim.x + threadIdx.x;
    if (i < total) eq[i] = (_Float16)__expf(q[i]);
}

// ---------------------------------------------------------------------------
// Kernel 2: scores[b, i] = log( Eq[b,:] . exp(keys[i,:]) )
// Block = 128 threads (4 waves). Each block covers 64 key rows; each wave
// covers 16 key rows x all 64 query rows via 4 stacked 16x16 WMMA tiles.
// Streaming: every key element is loaded once, exp'd, fed to WMMA.
// ---------------------------------------------------------------------------
__global__ void gemm_logsumexp_kernel(const float*   __restrict__ keys,   // [n][d] f32
                                      const _Float16* __restrict__ eq,    // [64][d] f16
                                      float*          __restrict__ scores,// [64][n] f32
                                      int n, int d) {
    const int lane  = threadIdx.x & 31;
    const int wave  = threadIdx.x >> 5;            // 0..3
    const int nbase = blockIdx.x * 64 + wave * 16; // this wave's 16 key rows
    const int half  = lane >> 4;                   // 0 or 1 (lane group)
    const int lmod  = lane & 15;

    const float* krow = keys + (size_t)(nbase + lmod) * d;

    v8f acc[4];
    #pragma unroll
    for (int m = 0; m < 4; ++m) acc[m] = (v8f){0.f,0.f,0.f,0.f,0.f,0.f,0.f,0.f};

    for (int kk = 0; kk < d; kk += 32) {
        // ---- B fragment (32x16 f16): lane = key column N, 16 consecutive K.
        // lanes 0-15 -> K = kk..kk+15 ; lanes 16-31 -> K = kk+16..kk+31
        const float4* kp4 = (const float4*)(krow + kk + half * 16);
        float4 f0 = kp4[0], f1 = kp4[1], f2 = kp4[2], f3 = kp4[3];
        float kv[16] = { f0.x, f0.y, f0.z, f0.w,
                         f1.x, f1.y, f1.z, f1.w,
                         f2.x, f2.y, f2.z, f2.w,
                         f3.x, f3.y, f3.z, f3.w };
        v16h bfrag;
        #pragma unroll
        for (int j = 0; j < 16; ++j) bfrag[j] = (_Float16)__expf(kv[j]);

        // ---- A fragments (16x32 f16) for the 4 query M-tiles.
        // lanes 0-15: elems 0..15 = K kk..kk+15
        // lanes 16-31: K kk+16..kk+31. Two aligned 16B loads per fragment.
        #pragma unroll
        for (int m = 0; m < 4; ++m) {
            const _Float16* arow = eq + (size_t)(m * 16 + lmod) * d + kk + half * 16;
            v8h lo = *(const v8h*)(arow);
            v8h hi = *(const v8h*)(arow + 8);
            v16h afrag;
            #pragma unroll
            for (int j = 0; j < 8; ++j) { afrag[j] = lo[j]; afrag[8 + j] = hi[j]; }
            acc[m] = __builtin_amdgcn_wmma_f32_16x16x32_f16(
                false, afrag, false, bfrag, (short)0, acc[m], false, false);
        }
    }

    // ---- Writeback: C/D layout: VGPR r, lanes 0-15 -> M=r, lanes 16-31 -> M=r+8
    const int col = nbase + lmod;
    #pragma unroll
    for (int m = 0; m < 4; ++m) {
        #pragma unroll
        for (int r = 0; r < 8; ++r) {
            int brow = m * 16 + r + half * 8;
            scores[(size_t)brow * n + col] = __logf(acc[m][r]);
        }
    }
}

// ---------------------------------------------------------------------------
// Kernel 3: per-query top-4 (jax tie-break: lower index wins), softmax over
// the 4 scores, gather theta rows and mag, write (theta | mag | w).
// One block (256 threads) per query row.
// ---------------------------------------------------------------------------
__device__ __forceinline__ bool better(float a, int ai, float b, int bi) {
    return (a > b) || (a == b && (unsigned)ai < (unsigned)bi);
}

__global__ void topk_kernel(const float* __restrict__ scores,  // [B][n]
                            const float* __restrict__ theta,   // [n][dh]
                            const float* __restrict__ mag,     // [n]
                            float* __restrict__ theta_out,     // [B][4][dh]
                            float* __restrict__ mag_out,       // [B][4]
                            float* __restrict__ w_out,         // [B][4]
                            int n, int dh) {
    const int b   = blockIdx.x;
    const int tid = threadIdx.x;
    const int NT  = 256;

    __shared__ float svals[NT * 4];
    __shared__ int   sidx[NT * 4];
    __shared__ int   best_i[4];
    __shared__ float best_w[4];

    const float* row = scores + (size_t)b * n;

    float v0 = NEG_INF, v1 = NEG_INF, v2 = NEG_INF, v3 = NEG_INF;
    int   i0 = -1, i1 = -1, i2 = -1, i3 = -1;

    for (int c = tid; c < n; c += NT) {
        float s = row[c];
        if (better(s, c, v3, i3)) {
            v3 = s; i3 = c;
            if (better(v3, i3, v2, i2)) {
                float tv = v2; int ti = i2; v2 = v3; i2 = i3; v3 = tv; i3 = ti;
                if (better(v2, i2, v1, i1)) {
                    tv = v1; ti = i1; v1 = v2; i1 = i2; v2 = tv; i2 = ti;
                    if (better(v1, i1, v0, i0)) {
                        tv = v0; ti = i0; v0 = v1; i0 = i1; v1 = tv; i1 = ti;
                    }
                }
            }
        }
    }
    svals[tid * 4 + 0] = v0; sidx[tid * 4 + 0] = i0;
    svals[tid * 4 + 1] = v1; sidx[tid * 4 + 1] = i1;
    svals[tid * 4 + 2] = v2; sidx[tid * 4 + 2] = i2;
    svals[tid * 4 + 3] = v3; sidx[tid * 4 + 3] = i3;
    __syncthreads();

    if (tid == 0) {
        float b0 = NEG_INF, b1 = NEG_INF, b2 = NEG_INF, b3 = NEG_INF;
        int   j0 = -1, j1 = -1, j2 = -1, j3 = -1;
        for (int c = 0; c < NT * 4; ++c) {
            float s = svals[c]; int id = sidx[c];
            if (id < 0) continue;
            if (better(s, id, b3, j3)) {
                b3 = s; j3 = id;
                if (better(b3, j3, b2, j2)) {
                    float tv = b2; int ti = j2; b2 = b3; j2 = j3; b3 = tv; j3 = ti;
                    if (better(b2, j2, b1, j1)) {
                        tv = b1; ti = j1; b1 = b2; j1 = j2; b2 = tv; j2 = ti;
                        if (better(b1, j1, b0, j0)) {
                            tv = b0; ti = j0; b0 = b1; j0 = j1; b1 = tv; j1 = ti;
                        }
                    }
                }
            }
        }
        // softmax over the top-4 scores
        float e0 = __expf(b0 - b0), e1 = __expf(b1 - b0),
              e2 = __expf(b2 - b0), e3 = __expf(b3 - b0);
        float inv = 1.0f / (e0 + e1 + e2 + e3);
        best_i[0] = j0; best_i[1] = j1; best_i[2] = j2; best_i[3] = j3;
        best_w[0] = e0 * inv; best_w[1] = e1 * inv;
        best_w[2] = e2 * inv; best_w[3] = e3 * inv;
    }
    __syncthreads();

    // gather theta rows (vectorized float4: dh=512 -> 128 float4 per row)
    const int dh4 = dh >> 2;
    #pragma unroll
    for (int j = 0; j < 4; ++j) {
        const float4* trow = (const float4*)(theta + (size_t)best_i[j] * dh);
        float4*       orow = (float4*)(theta_out + ((size_t)b * 4 + j) * dh);
        for (int c = tid; c < dh4; c += NT) orow[c] = trow[c];
    }
    if (tid < 4) {
        mag_out[b * 4 + tid] = mag[best_i[tid]];
        w_out[b * 4 + tid]   = best_w[tid];
    }
}

// ---------------------------------------------------------------------------
extern "C" void kernel_launch(void* const* d_in, const int* in_sizes, int n_in,
                              void* d_out, int out_size, void* d_ws, size_t ws_size,
                              hipStream_t stream) {
    const float* query = (const float*)d_in[0];   // [B, d]
    const float* keys  = (const float*)d_in[1];   // [n, d]
    const float* theta = (const float*)d_in[2];   // [n, d/2]
    const float* mag   = (const float*)d_in[3];   // [n, 1]
    // top_k = 4 (compile-time, per reference)

    const int n  = in_sizes[3];           // 8192
    const int d  = in_sizes[1] / n;       // 1024
    const int B  = in_sizes[0] / d;       // 64
    const int dh = in_sizes[2] / n;       // 512

    // workspace layout: Eq f16 [B*d], then scores f32 [B*n]
    _Float16* eq = (_Float16*)d_ws;
    size_t eq_bytes = ((size_t)B * d * sizeof(_Float16) + 255) & ~(size_t)255;
    float* scores = (float*)((char*)d_ws + eq_bytes);

    float* theta_out = (float*)d_out;
    float* mag_out   = theta_out + (size_t)B * 4 * dh;
    float* w_out     = mag_out + (size_t)B * 4;

    // 1) exp(query) -> f16
    {
        int total = B * d;
        expq_kernel<<<(total + 255) / 256, 256, 0, stream>>>(query, eq, total);
    }
    // 2) streaming fused exp + WMMA GEMM + log -> scores
    gemm_logsumexp_kernel<<<n / 64, 128, 0, stream>>>(keys, eq, scores, n, d);
    // 3) top-4 + softmax + gather
    topk_kernel<<<B, 256, 0, stream>>>(scores, theta, mag,
                                       theta_out, mag_out, w_out, n, dh);
}
